// GAT1_44830868636064
// MI455X (gfx1250) — matrix operand order repaired and checked by reading, not compiled, not run on hardware
//
#include <hip/hip_runtime.h>
#include <hip/hip_bf16.h>
#include <math.h>

typedef __attribute__((ext_vector_type(16))) _Float16 v16h;
typedef __attribute__((ext_vector_type(8)))  float    v8f;
typedef __attribute__((ext_vector_type(4)))  int      i32x4;

#define Nn 8192      // nodes
#define Ff 64        // features per layer
#define FP 80        // padded GEMM N-dim: 64 feats + 1 denom col + 15 zero
#define Gg 64        // graphs
#define KSPLIT 8
#define KCH (Nn / KSPLIT)

// 0/1 int -> packed pair of f16 (1.0 = 0x3C00), exact.
__device__ __forceinline__ unsigned pack01(int a, int b) {
    return (a ? 0x3C00u : 0u) | ((b ? 0x3C00u : 0u) << 16);
}

// h[j,f] = sum_d x[j,d] * W[d,f] + b[f]   (tiny GEMM, fp32 VALU is plenty)
__global__ void k_linear(const float* __restrict__ x, int ldx, int Din,
                         const float* __restrict__ W, const float* __restrict__ b,
                         float* __restrict__ h) {
    int idx = blockIdx.x * blockDim.x + threadIdx.x;
    if (idx >= Nn * Ff) return;
    int j = idx >> 6, f = idx & 63;
    const float* xr = x + (size_t)j * ldx;
    float acc = b[f];
    for (int d = 0; d < Din; ++d) acc = fmaf(xr[d], W[d * Ff + f], acc);
    h[idx] = acc;
}

// s2[j] = h[j,:] . aw + ab
__global__ void k_s2(const float* __restrict__ h, const float* __restrict__ aw,
                     const float* __restrict__ ab, float* __restrict__ s2) {
    int j = blockIdx.x * blockDim.x + threadIdx.x;
    if (j >= Nn) return;
    const float* hr = h + (size_t)j * Ff;
    float acc = ab[0];
#pragma unroll
    for (int f = 0; f < Ff; ++f) acc = fmaf(hr[f], aw[f], acc);
    s2[j] = acc;
}

// global max of s2 (softmax shift)
__global__ void k_max(const float* __restrict__ s2, float* __restrict__ mout) {
    __shared__ float red[256];
    float m = -1e30f;
    for (int j = threadIdx.x; j < Nn; j += 256) m = fmaxf(m, s2[j]);
    red[threadIdx.x] = m;
    __syncthreads();
    for (int s = 128; s > 0; s >>= 1) {
        if (threadIdx.x < s) red[threadIdx.x] = fmaxf(red[threadIdx.x], red[threadIdx.x + s]);
        __syncthreads();
    }
    if (threadIdx.x == 0) mout[0] = red[0];
}

// Build feature-major B panel: whT[f][j] = w[j]*h[j,f] (f<64), w[j] (f==64), 0 pad.
// Split into hi/lo f16 so (hi+lo) recovers ~fp32 precision in the WMMA.
__global__ void k_build(const float* __restrict__ h, const float* __restrict__ s2,
                        const float* __restrict__ mmax,
                        _Float16* __restrict__ whT_hi, _Float16* __restrict__ whT_lo) {
    int idx = blockIdx.x * blockDim.x + threadIdx.x;
    if (idx >= Nn * FP) return;
    int f = idx / Nn, j = idx - f * Nn;
    float w = expf(s2[j] - mmax[0]);
    float v = (f < Ff) ? w * h[(size_t)j * Ff + f] : (f == Ff ? w : 0.0f);
    _Float16 hi = (_Float16)v;
    whT_hi[idx] = hi;
    whT_lo[idx] = (_Float16)(v - (float)hi);
}

// Fused masked-GEMM: num_part[p][i][0:80] = sum_{j in Kslice p} mask[i,j]*[w*h|w]
// A = 0/1 mask converted to exact f16 tiles, B = whT_hi + whT_lo, f32 accum WMMA.
// sched_group_barrier pins the pipeline per k-chunk: 24 VMEM reads issued
// together -> A-pack VALU -> 10 back-to-back WMMAs. This forces the register
// allocator to keep all B fragments live (partial s_wait_loadcnt instead of
// ten serialized full-latency waits).
__global__ void __launch_bounds__(256) k_attn(const int* __restrict__ adj,
                                              const _Float16* __restrict__ whT_hi,
                                              const _Float16* __restrict__ whT_lo,
                                              float* __restrict__ num_part) {
    const int lane  = threadIdx.x & 31;
    const int wave  = threadIdx.x >> 5;     // K-split id (0..7)
    const int row   = lane & 15;            // A: M row / B: N col / C: N col
    const int half  = lane >> 4;            // lane-half selector
    const int mbase = blockIdx.x * 16;
    const int kstart = wave * KCH;

    const int* arow = adj + (size_t)(mbase + row) * Nn;

    v8f zero = {0.f,0.f,0.f,0.f,0.f,0.f,0.f,0.f};
    v8f acc[5];
#pragma unroll
    for (int t = 0; t < 5; ++t) acc[t] = zero;

    for (int k0 = kstart; k0 < kstart + KCH; k0 += 32) {
        // ---- A loads first (16 contiguous ints/lane, K[k0+16*half .. +16), matching B)
        const int kA = k0 + half * 16;
        i32x4 a0 = *(const i32x4*)(arow + kA);
        i32x4 a1 = *(const i32x4*)(arow + kA + 4);
        i32x4 a2 = *(const i32x4*)(arow + kA + 8);
        i32x4 a3 = *(const i32x4*)(arow + kA + 12);

        // ---- all B fragments hoisted: 32B contiguous per lane (contiguous k order, same as A:
        // lanes 0-15 K[k0..k0+16), lanes 16-31 K[k0+16..k0+32)), consumption order
        const int jB = k0 + half * 16;
        v16h bh[5], bl[5];
#pragma unroll
        for (int ft = 0; ft < 5; ++ft) {
            bh[ft] = *(const v16h*)(whT_hi + (size_t)(ft * 16 + row) * Nn + jB);
            bl[ft] = *(const v16h*)(whT_lo + (size_t)(ft * 16 + row) * Nn + jB);
        }
        __builtin_prefetch(arow + k0 + 128, 0, 1);   // mask stream, 4 iters ahead

        // ---- pack 0/1 ints -> exact f16 A fragment (overlaps B loads in flight)
        union { v16h h; unsigned u[8]; } A;
        A.u[0] = pack01(a0[0], a0[1]); A.u[1] = pack01(a0[2], a0[3]);
        A.u[2] = pack01(a1[0], a1[1]); A.u[3] = pack01(a1[2], a1[3]);
        A.u[4] = pack01(a2[0], a2[1]); A.u[5] = pack01(a2[2], a2[3]);
        A.u[6] = pack01(a3[0], a3[1]); A.u[7] = pack01(a3[2], a3[3]);

#pragma unroll
        for (int ft = 0; ft < 5; ++ft) {
            acc[ft] = __builtin_amdgcn_wmma_f32_16x16x32_f16(
                false, A.h, false, bh[ft], (short)0, acc[ft], false, false);
            acc[ft] = __builtin_amdgcn_wmma_f32_16x16x32_f16(
                false, A.h, false, bl[ft], (short)0, acc[ft], false, false);
        }

#if __has_builtin(__builtin_amdgcn_sched_group_barrier)
        // Pipeline shape for this iteration: loads | pack | matrix ops.
        __builtin_amdgcn_sched_group_barrier(0x0020, 24, 0); // VMEM reads
        __builtin_amdgcn_sched_group_barrier(0x0002, 56, 0); // VALU (cmp/sel/or pack)
        __builtin_amdgcn_sched_group_barrier(0x0008, 10, 0); // WMMA
#endif
    }

    // C/D layout: VGPR v -> row v (lanes 0-15) / row v+8 (lanes 16-31), col = lane&15
    float* base = num_part + ((size_t)wave * Nn + mbase) * FP;
#pragma unroll
    for (int ft = 0; ft < 5; ++ft)
#pragma unroll
        for (int v = 0; v < 8; ++v)
            base[(size_t)(v + half * 8) * FP + ft * 16 + row] = acc[ft][v];
}

// Deterministic K-split reduction + softmax normalize + ReLU -> xcat[:, col0:col0+64]
__global__ void k_finish(const float* __restrict__ num_part, float* __restrict__ xcat, int col0) {
    int idx = blockIdx.x * blockDim.x + threadIdx.x;
    if (idx >= Nn * Ff) return;
    int i = idx >> 6, f = idx & 63;
    float num = 0.f, den = 0.f;
#pragma unroll
    for (int p = 0; p < KSPLIT; ++p) {
        const float* r = num_part + ((size_t)p * Nn + i) * FP;
        num += r[f];
        den += r[Ff];
    }
    float v = num / den;
    xcat[(size_t)i * 192 + col0 + f] = v > 0.f ? v : 0.f;
}

// Segment mean-pool: batch is sorted, binary search the [g,g+1) range.
__global__ void k_pool(const float* __restrict__ xcat, const int* __restrict__ batch,
                       float* __restrict__ pooled) {
    __shared__ int bnds[2];
    int g = blockIdx.x;
    if (threadIdx.x == 0) {
        int lo = 0, hi = Nn;
        while (lo < hi) { int mid = (lo + hi) >> 1; if (batch[mid] < g) lo = mid + 1; else hi = mid; }
        bnds[0] = lo;
        hi = Nn;
        while (lo < hi) { int mid = (lo + hi) >> 1; if (batch[mid] <= g) lo = mid + 1; else hi = mid; }
        bnds[1] = lo;
    }
    __syncthreads();
    int s = bnds[0], e = bnds[1];
    float acc = 0.f;
    for (int r = s; r < e; ++r) acc += xcat[(size_t)r * 192 + threadIdx.x];
    pooled[g * 192 + threadIdx.x] = acc / fmaxf((float)(e - s), 1.0f);
}

// Classifier head + softmax, one thread per graph.
__global__ void k_head(const float* __restrict__ pooled, const float* __restrict__ Wo,
                       const float* __restrict__ bo, float* __restrict__ out) {
    int g = blockIdx.x * blockDim.x + threadIdx.x;
    if (g >= Gg) return;
    float logits[10], mx = -1e30f;
    for (int c = 0; c < 10; ++c) {
        float acc = bo[c];
        for (int r = 0; r < 192; ++r) acc = fmaf(pooled[g * 192 + r], Wo[r * 10 + c], acc);
        logits[c] = acc;
        mx = fmaxf(mx, acc);
    }
    float sum = 0.f;
    for (int c = 0; c < 10; ++c) { logits[c] = expf(logits[c] - mx); sum += logits[c]; }
    for (int c = 0; c < 10; ++c) out[g * 10 + c] = logits[c] / sum;
}

extern "C" void kernel_launch(void* const* d_in, const int* in_sizes, int n_in,
                              void* d_out, int out_size, void* d_ws, size_t ws_size,
                              hipStream_t stream) {
    const float* features = (const float*)d_in[0];
    const int*   adj      = (const int*)d_in[1];
    const int*   batch    = (const int*)d_in[2];
    const float* Wl[3] = {(const float*)d_in[3],  (const float*)d_in[9],  (const float*)d_in[15]};
    const float* bl[3] = {(const float*)d_in[4],  (const float*)d_in[10], (const float*)d_in[16]};
    // s1 (a*1w/a*1b) cancels inside the row softmax -> only the a*2 vectors matter.
    const float* aw[3] = {(const float*)d_in[7],  (const float*)d_in[13], (const float*)d_in[19]};
    const float* ab[3] = {(const float*)d_in[8],  (const float*)d_in[14], (const float*)d_in[20]};
    const float* Wo = (const float*)d_in[21];
    const float* bo = (const float*)d_in[22];
    float* out = (float*)d_out;

    char* ws = (char*)d_ws;
    size_t off = 0;
    auto alloc = [&](size_t bytes) -> void* {
        off = (off + 255) & ~(size_t)255;
        void* p = ws + off;
        off += bytes;
        return p;
    };
    float*    h        = (float*)alloc((size_t)Nn * Ff * 4);
    float*    s2       = (float*)alloc((size_t)Nn * 4);
    float*    mmax     = (float*)alloc(4);
    _Float16* whT_hi   = (_Float16*)alloc((size_t)FP * Nn * 2);
    _Float16* whT_lo   = (_Float16*)alloc((size_t)FP * Nn * 2);
    float*    num_part = (float*)alloc((size_t)KSPLIT * Nn * FP * 4);
    float*    xcat     = (float*)alloc((size_t)Nn * 192 * 4);
    float*    pooled   = (float*)alloc((size_t)Gg * 192 * 4);

    for (int l = 0; l < 3; ++l) {
        const float* xin = (l == 0) ? features : (xcat + (size_t)(l - 1) * 64);
        int ldx = (l == 0) ? 128 : 192;
        int Din = (l == 0) ? 128 : 64;
        k_linear<<<(Nn * Ff + 255) / 256, 256, 0, stream>>>(xin, ldx, Din, Wl[l], bl[l], h);
        k_s2<<<(Nn + 255) / 256, 256, 0, stream>>>(h, aw[l], ab[l], s2);
        k_max<<<1, 256, 0, stream>>>(s2, mmax);
        k_build<<<(Nn * FP + 255) / 256, 256, 0, stream>>>(h, s2, mmax, whT_hi, whT_lo);
        k_attn<<<Nn / 16, 256, 0, stream>>>(adj, whT_hi, whT_lo, num_part);
        k_finish<<<(Nn * Ff + 255) / 256, 256, 0, stream>>>(num_part, xcat, l * 64);
    }
    k_pool<<<Gg, 192, 0, stream>>>(xcat, batch, pooled);
    k_head<<<1, 64, 0, stream>>>(pooled, Wo, bo, out);
}
